// MambaByteLayer_10075993276740
// MI455X (gfx1250) — matrix operand order repaired and checked by reading, not compiled, not run on hardware
//
#include <hip/hip_runtime.h>
#include <hip/hip_bf16.h>

// ---------------------------------------------------------------------------
// Problem constants (B, L, D) = (2, 1024, 1024); E=2048, N=16, R=64, CONV=4, K=2
// ---------------------------------------------------------------------------
#define Ld   1024
#define Dd   1024
#define Ee   2048
#define Nn   16
#define Rr   64
#define Bb   2
#define Kk   2

// ---------------------------------------------------------------------------
// Optional gfx1250 async global->LDS path (probe-guarded)
// ---------------------------------------------------------------------------
#if defined(__has_builtin)
#if __has_builtin(__builtin_amdgcn_global_load_async_to_lds_b128) && \
    __has_builtin(__builtin_amdgcn_s_wait_asynccnt)
#define HAVE_ASYNC_LDS 1
#endif
#endif
#ifndef HAVE_ASYNC_LDS
#define HAVE_ASYNC_LDS 0
#endif

typedef __attribute__((ext_vector_type(4))) int v4i_t;
typedef __attribute__((address_space(1))) v4i_t* gv4i_t;   // global int4*
typedef __attribute__((address_space(3))) v4i_t* lv4i_t;   // LDS int4*

// ---------------------------------------------------------------------------
// CDNA5 WMMA types
// ---------------------------------------------------------------------------
typedef __attribute__((ext_vector_type(16))) __bf16 v16bf;
typedef __attribute__((ext_vector_type(8)))  float  v8f;

union FragBF {
  v16bf v;
  unsigned short s[16];
  uint4 q[2];
};

__device__ __forceinline__ unsigned short f2bf(float f) {
  unsigned u = __float_as_uint(f);
  u = u + 0x7FFFu + ((u >> 16) & 1u);   // round-to-nearest-even truncation
  return (unsigned short)(u >> 16);
}

__device__ __forceinline__ ushort4 pack4(float4 v) {
  ushort4 r;
  r.x = f2bf(v.x); r.y = f2bf(v.y); r.z = f2bf(v.z); r.w = f2bf(v.w);
  return r;
}

enum { A_NONE = 0, A_SILU = 1, A_GELU = 2, A_SIGMOID = 3, A_SOFTPLUS = 4 };

__device__ __forceinline__ float act_apply(float x, int act) {
  switch (act) {
    case A_SILU:    return x / (1.f + __expf(-x));
    case A_GELU: {  // tanh approximation (jax.nn.gelu default)
      float x3 = x * x * x;
      return 0.5f * x * (1.f + tanhf(0.7978845608028654f * (x + 0.044715f * x3)));
    }
    case A_SIGMOID: return 1.f / (1.f + __expf(-x));
    case A_SOFTPLUS:return (x > 20.f) ? x : log1pf(__expf(x));
    default:        return x;
  }
}

// ---------------------------------------------------------------------------
// bf16-WMMA GEMM:  C[M,N] = act(A[M,K] * B[K,N] + bias[N])
//   Block tile 128(M) x 64(N), KTILE 64, 256 threads = 8 waves (4x2).
//   Each wave: 32x32 accumulator quad -> 8 v_wmma per K-tile.
//   f32 tiles staged to LDS as bf16 via b128 loads + packed b64 stores.
//   transA: A stored K-major (A[k*lda+m]); transposed once into LDS.
// Requirements (all call sites satisfy): M % 128 == 0 when transA, K % 64 == 0.
// ---------------------------------------------------------------------------
__global__ __launch_bounds__(256) void gemm_wmma_bf16(
    const float* __restrict__ A, const float* __restrict__ Bm,
    const float* __restrict__ bias, float* __restrict__ C,
    int M, int N, int Kd, int lda, int ldb, int ldc, int transA, int act)
{
  __shared__ __align__(16) unsigned short a_lds[128][72];  // rows: M, cols: K (144B stride)
  __shared__ __align__(16) unsigned short b_lds[64][72];   // rows: K, cols: N

  const int tid  = threadIdx.x;
  const int wave = tid >> 5;
  const int lane = tid & 31;
  const int half = lane >> 4;
  const int l16  = lane & 15;
  const int waveM = wave >> 1;        // 0..3
  const int waveN = wave & 1;         // 0..1

  const int m0 = blockIdx.y * 128;
  const int n0 = blockIdx.x * 64;

  v8f acc[2][2];
#pragma unroll
  for (int i = 0; i < 2; ++i)
#pragma unroll
    for (int j = 0; j < 2; ++j)
      acc[i][j] = (v8f){0.f, 0.f, 0.f, 0.f, 0.f, 0.f, 0.f, 0.f};

  const int tcol = tid & 15;   // 16 column groups of 4
  const int trow = tid >> 4;   // 16 rows

  for (int k0 = 0; k0 < Kd; k0 += 64) {
    // ---- stage A tile (128 x 64) into LDS as bf16 ----
    if (!transA) {
#pragma unroll
      for (int it = 0; it < 8; ++it) {
        int row = trow + it * 16;          // 0..127
        int col = tcol * 4;                // 0..60
        int gm  = m0 + row;
        float4 v = {0.f, 0.f, 0.f, 0.f};
        if (gm < M) {
          const float* ap = A + (size_t)gm * lda + k0 + col;
          v = *reinterpret_cast<const float4*>(ap);
          __builtin_prefetch(ap + 64, 0, 3);     // next K-tile
        }
        *reinterpret_cast<ushort4*>(&a_lds[row][col]) = pack4(v);
      }
    } else {
      // A[k][m] -> a_lds[m][k]
#pragma unroll
      for (int it = 0; it < 4; ++it) {
        int krow = trow + it * 16;         // 0..63
        int mc   = tcol * 8;               // 0..120
        const float* ap = A + (size_t)(k0 + krow) * lda + m0 + mc;
        if (m0 + mc + 7 < M) {
          float4 v0 = *reinterpret_cast<const float4*>(ap);
          float4 v1 = *reinterpret_cast<const float4*>(ap + 4);
          a_lds[mc + 0][krow] = f2bf(v0.x);
          a_lds[mc + 1][krow] = f2bf(v0.y);
          a_lds[mc + 2][krow] = f2bf(v0.z);
          a_lds[mc + 3][krow] = f2bf(v0.w);
          a_lds[mc + 4][krow] = f2bf(v1.x);
          a_lds[mc + 5][krow] = f2bf(v1.y);
          a_lds[mc + 6][krow] = f2bf(v1.z);
          a_lds[mc + 7][krow] = f2bf(v1.w);
        } else {
#pragma unroll
          for (int j = 0; j < 8; ++j) a_lds[mc + j][krow] = (unsigned short)0;
        }
      }
    }
    // ---- stage B tile (64 x 64) into LDS as bf16 ----
#pragma unroll
    for (int it = 0; it < 4; ++it) {
      int row = trow + it * 16;            // 0..63 (K)
      int col = tcol * 4;                  // 0..60 (N)
      int gn  = n0 + col;
      const float* bp = Bm + (size_t)(k0 + row) * ldb + gn;
      float4 v = {0.f, 0.f, 0.f, 0.f};
      if (gn + 3 < N) {
        v = *reinterpret_cast<const float4*>(bp);
        __builtin_prefetch(bp + (size_t)64 * ldb, 0, 3);
      } else {
        if (gn + 0 < N) v.x = bp[0];
        if (gn + 1 < N) v.y = bp[1];
        if (gn + 2 < N) v.z = bp[2];
        if (gn + 3 < N) v.w = bp[3];
      }
      *reinterpret_cast<ushort4*>(&b_lds[row][col]) = pack4(v);
    }
    __syncthreads();

    // ---- 2 K-steps of 32, 4 WMMAs each ----
#pragma unroll
    for (int kk = 0; kk < 64; kk += 32) {
      FragBF fa[2], fb[2];
#pragma unroll
      for (int ms = 0; ms < 2; ++ms) {
        const unsigned short* ar = &a_lds[waveM * 32 + ms * 16 + l16][0];
        fa[ms].q[0] = *reinterpret_cast<const uint4*>(ar + kk + half * 8);
        fa[ms].q[1] = *reinterpret_cast<const uint4*>(ar + kk + half * 8 + 16);
      }
#pragma unroll
      for (int ns = 0; ns < 2; ++ns) {
        const unsigned short* br = &b_lds[kk + half * 8][waveN * 32 + ns * 16 + l16];
#pragma unroll
        for (int s = 0; s < 8; ++s) {
          fb[ns].s[s]     = br[s * 72];
          fb[ns].s[s + 8] = br[(s + 16) * 72];
        }
      }
#pragma unroll
      for (int ms = 0; ms < 2; ++ms)
#pragma unroll
        for (int ns = 0; ns < 2; ++ns)
          acc[ms][ns] = __builtin_amdgcn_wmma_f32_16x16x32_bf16(
              false, fa[ms].v, false, fb[ns].v, (short)0, acc[ms][ns],
              false, false);
    }
    __syncthreads();
  }

  // ---- store (C/D layout: VGPR v -> M = v + 8*half, N = lane&15) ----
#pragma unroll
  for (int ms = 0; ms < 2; ++ms)
#pragma unroll
    for (int ns = 0; ns < 2; ++ns) {
      int rowb = m0 + waveM * 32 + ms * 16;
      int col  = n0 + waveN * 32 + ns * 16 + l16;
#pragma unroll
      for (int v = 0; v < 8; ++v) {
        int row = rowb + v + 8 * half;
        if (row < M && col < N) {
          float x = acc[ms][ns][v];
          if (bias) x += bias[col];
          C[(size_t)row * ldc + col] = act_apply(x, act);
        }
      }
    }
}

// ---------------------------------------------------------------------------
// bn_w ternary quantization (deterministic 2-stage reduction)
// ---------------------------------------------------------------------------
__global__ __launch_bounds__(256) void bn_partial_kernel(
    const float* __restrict__ w, float* __restrict__ part)
{
  __shared__ float red[256];
  int blk  = blockIdx.x;          // 0 .. K*64-1
  int k    = blk >> 6;
  int p    = blk & 63;
  size_t base = (size_t)k * (Ld * Ld) + (size_t)p * 16384;
  float s = 0.f;
  for (int i = threadIdx.x; i < 16384; i += 256) s += fabsf(w[base + i]);
  red[threadIdx.x] = s;
  __syncthreads();
  for (int off = 128; off > 0; off >>= 1) {
    if (threadIdx.x < off) red[threadIdx.x] += red[threadIdx.x + off];
    __syncthreads();
  }
  if (threadIdx.x == 0) part[blk] = red[0];
}

__global__ void bn_scale_kernel(const float* __restrict__ part,
                                float* __restrict__ scale)
{
  __shared__ float red[64];
  int k = blockIdx.x;
  red[threadIdx.x] = part[k * 64 + threadIdx.x];
  __syncthreads();
  for (int off = 32; off > 0; off >>= 1) {
    if (threadIdx.x < off) red[threadIdx.x] += red[threadIdx.x + off];
    __syncthreads();
  }
  if (threadIdx.x == 0)
    scale[k] = red[0] / (float)(Ld * Ld) + 1e-6f;
}

__global__ void bn_quant_kernel(const float* __restrict__ w,
                                const float* __restrict__ scale,
                                float* __restrict__ wq)
{
  size_t i = (size_t)blockIdx.x * 256 + threadIdx.x;
  int k = (int)(i >> 20);                 // L*L = 2^20
  float s = scale[k];
  float q = rintf(w[i] / s);
  q = fminf(1.f, fmaxf(-1.f, q));
  wq[i] = q * s;
}

// ---------------------------------------------------------------------------
// Depthwise causal conv (CONV=4) + bias + silu. Input: xz[b,l,0:E] part.
// ---------------------------------------------------------------------------
__global__ void conv_silu_kernel(const float* __restrict__ xz,
                                 const float* __restrict__ cw,
                                 const float* __restrict__ cb,
                                 float* __restrict__ xh)
{
  size_t idx = (size_t)blockIdx.x * 256 + threadIdx.x;   // over b*L*E
  int e = (int)(idx & (Ee - 1));
  size_t bl = idx >> 11;                                  // bi*L + l
  int l = (int)(bl & (Ld - 1));
  float acc = cb[e];
#pragma unroll
  for (int j = 0; j < 4; ++j) {
    int lp = l - 3 + j;
    if (lp >= 0) acc += cw[e * 4 + j] * xz[(bl - 3 + j) * (2 * Ee) + e];
  }
  xh[bl * Ee + e] = acc / (1.f + __expf(-acc));
}

// ---------------------------------------------------------------------------
// Mamba selective scan. One thread per (bi, e); 16-wide state in registers.
// Emits  ymb = (y + D_skip*xh) * silu(z).
// ---------------------------------------------------------------------------
__global__ __launch_bounds__(256) void mamba_scan_kernel(
    const float* __restrict__ xz,  const float* __restrict__ xh,
    const float* __restrict__ dt,  const float* __restrict__ dbc,
    const float* __restrict__ A_log, const float* __restrict__ D_skip,
    float* __restrict__ ymb)
{
  __shared__ float bc[2 * Nn];
  int e  = blockIdx.x * 256 + threadIdx.x;
  int bi = blockIdx.y;
  float an[Nn], h[Nn];
#pragma unroll
  for (int n = 0; n < Nn; ++n) {
    an[n] = -__expf(A_log[e * Nn + n]);
    h[n]  = 0.f;
  }
  float dsk = D_skip[e];
  for (int l = 0; l < Ld; ++l) {
    size_t row = (size_t)bi * Ld + l;
    if (threadIdx.x < 2 * Nn)
      bc[threadIdx.x] = dbc[row * (Rr + 2 * Nn) + Rr + threadIdx.x];
    __syncthreads();
    float dtv = dt[row * Ee + e];
    float xv  = xh[row * Ee + e];
    float zv  = xz[row * (2 * Ee) + Ee + e];
    float dx  = dtv * xv;
    float y = 0.f;
#pragma unroll
    for (int n = 0; n < Nn; ++n) {
      h[n] = __expf(dtv * an[n]) * h[n] + dx * bc[n];
      y += h[n] * bc[Nn + n];
    }
    float sz = zv / (1.f + __expf(-zv));
    ymb[row * Ee + e] = (y + dsk * xv) * sz;
    __syncthreads();
  }
}

// ---------------------------------------------------------------------------
// Liquid-NN scan: h += dt*(-h/tau + tanh(xw_l + h @ wh)). One block per k.
// ---------------------------------------------------------------------------
__global__ __launch_bounds__(1024) void liquid_scan_kernel(
    const float* __restrict__ xw, const float* __restrict__ wh,
    const float* __restrict__ tau, float* __restrict__ hs)
{
  __shared__ float h[Bb * Dd];
  int k = blockIdx.x;
  int tid = threadIdx.x;
  h[tid] = 0.f; h[tid + 1024] = 0.f;
  __syncthreads();
  const float* whk = wh + (size_t)k * Dd * Dd;
  const float* tk  = tau + (size_t)k * Dd;
  for (int l = 0; l < Ld; ++l) {
    float outv[2];
#pragma unroll
    for (int p = 0; p < 2; ++p) {
      int o  = tid + p * 1024;
      int bi = o >> 10;
      int dc = o & 1023;
      float acc = xw[(((size_t)(k * Bb + bi)) * Ld + l) * Dd + dc];
      const float* hp = h + bi * Dd;
      for (int d = 0; d < Dd; ++d) acc += hp[d] * whk[(size_t)d * Dd + dc];
      float hv = h[o];
      outv[p] = hv + 0.1f * (-hv / tk[dc] + tanhf(acc));
    }
    __syncthreads();
#pragma unroll
    for (int p = 0; p < 2; ++p) {
      int o  = tid + p * 1024;
      int bi = o >> 10;
      int dc = o & 1023;
      h[o] = outv[p];
      hs[(((size_t)(k * Bb + bi)) * Ld + l) * Dd + dc] = outv[p];
    }
    __syncthreads();
  }
}

// ---------------------------------------------------------------------------
// Exact top-k(256) threshold per row of 1024 via binary search on uint bits.
// Row staged in LDS via gfx1250 async global->LDS copy when available.
// Writes xm = x * (|x| >= thr).
// ---------------------------------------------------------------------------
__global__ __launch_bounds__(256) void topk_mask_kernel(
    const float* __restrict__ xin, float* __restrict__ xm)
{
  __shared__ __align__(16) float rowf[Dd];
  __shared__ int red[256];
  const float* row = xin + (size_t)blockIdx.x * Dd;
  float* orow = xm + (size_t)blockIdx.x * Dd;
  int tid = threadIdx.x;
#if HAVE_ASYNC_LDS
  __builtin_amdgcn_global_load_async_to_lds_b128(
      (gv4i_t)(row + tid * 4),
      (lv4i_t)(rowf + tid * 4), 0, 0);
  __builtin_amdgcn_s_wait_asynccnt(0);
#else
  *reinterpret_cast<float4*>(&rowf[tid * 4]) =
      *reinterpret_cast<const float4*>(row + tid * 4);
#endif
  __syncthreads();

  unsigned lo = 0u, hi = 0x7F800001u;   // cnt(lo) == 1024 >= 256
  while (hi - lo > 1u) {
    unsigned mid = lo + ((hi - lo) >> 1);
    int c = 0;
#pragma unroll
    for (int j = 0; j < 4; ++j)
      c += (__float_as_uint(fabsf(rowf[tid + j * 256])) >= mid) ? 1 : 0;
    red[tid] = c;
    __syncthreads();
    for (int off = 128; off > 0; off >>= 1) {
      if (tid < off) red[tid] += red[tid + off];
      __syncthreads();
    }
    int total = red[0];
    __syncthreads();
    if (total >= (Dd / 4)) lo = mid; else hi = mid;
  }
  float thr = __uint_as_float(lo);
#pragma unroll
  for (int j = 0; j < 4; ++j) {
    int i = tid + j * 256;
    float v = rowf[i];
    orow[i] = (fabsf(v) >= thr) ? v : 0.f;
  }
}

// ---------------------------------------------------------------------------
// act_quant per row of 1024: s = max|x|+1e-6; q = rint(clip(x/s*127))*s/127
// ---------------------------------------------------------------------------
__global__ __launch_bounds__(256) void actq_kernel(
    const float* __restrict__ xin, float* __restrict__ q)
{
  __shared__ float red[256];
  const float* row = xin + (size_t)blockIdx.x * Dd;
  float* orow = q + (size_t)blockIdx.x * Dd;
  int tid = threadIdx.x;
  float m = 0.f;
#pragma unroll
  for (int j = 0; j < 4; ++j) m = fmaxf(m, fabsf(row[tid + j * 256]));
  red[tid] = m;
  __syncthreads();
  for (int off = 128; off > 0; off >>= 1) {
    if (tid < off) red[tid] = fmaxf(red[tid], red[tid + off]);
    __syncthreads();
  }
  float s = red[0] + 1e-6f;
  float inv = 127.f / s, outs = s / 127.f;
#pragma unroll
  for (int j = 0; j < 4; ++j) {
    int i = tid + j * 256;
    float v = row[i] * inv;
    v = fminf(127.f, fmaxf(-127.f, v));
    orow[i] = rintf(v) * outs;
  }
}

// ---------------------------------------------------------------------------
// x = aq*g + v*(1-g)   (g overwritten in place)
// ---------------------------------------------------------------------------
__global__ void combine_kernel(const float* __restrict__ aq,
                               float* __restrict__ g,
                               const float* __restrict__ v)
{
  size_t i = (size_t)blockIdx.x * 256 + threadIdx.x;
  float gv = g[i];
  g[i] = aq[i] * gv + v[i] * (1.f - gv);
}

// out[k,b,l,d] += residual[b,l,d]
__global__ void residual_add_kernel(float* __restrict__ out,
                                    const float* __restrict__ res)
{
  size_t i = (size_t)blockIdx.x * 256 + threadIdx.x;
  out[i] += res[i & ((size_t)Bb * Ld * Dd - 1)];
}

// ---------------------------------------------------------------------------
// Host-side helpers
// ---------------------------------------------------------------------------
static inline void gemm(const float* A, const float* Bm, const float* bias,
                        float* C, int M, int N, int Kd,
                        int lda, int ldb, int ldc, int transA, int act,
                        hipStream_t s)
{
  dim3 grid((N + 63) / 64, (M + 127) / 128);
  gemm_wmma_bf16<<<grid, 256, 0, s>>>(A, Bm, bias, C, M, N, Kd,
                                      lda, ldb, ldc, transA, act);
}

static void run_mamba(const float* xin, int b, const float* const* P,
                      float* xz, float* xh, float* dbc, float* dt,
                      float* ymb, float* out, hipStream_t s)
{
  const float *in_proj = P[0], *conv_w = P[1], *conv_b = P[2], *x_proj = P[3],
              *dt_w = P[4], *dt_b = P[5], *A_log = P[6], *D_skip = P[7],
              *out_proj = P[8];
  int M = b * Ld;
  gemm(xin, in_proj, nullptr, xz, M, 2 * Ee, Dd, Dd, 2 * Ee, 2 * Ee, 0, A_NONE, s);
  conv_silu_kernel<<<(size_t)M * Ee / 256, 256, 0, s>>>(xz, conv_w, conv_b, xh);
  gemm(xh, x_proj, nullptr, dbc, M, Rr + 2 * Nn, Ee, Ee, Rr + 2 * Nn, Rr + 2 * Nn,
       0, A_NONE, s);
  gemm(dbc, dt_w, dt_b, dt, M, Ee, Rr, Rr + 2 * Nn, Ee, Ee, 0, A_SOFTPLUS, s);
  mamba_scan_kernel<<<dim3(Ee / 256, b), 256, 0, s>>>(xz, xh, dt, dbc, A_log,
                                                      D_skip, ymb);
  gemm(ymb, out_proj, nullptr, out, M, Dd, Ee, Ee, Dd, Dd, 0, A_NONE, s);
}

// Workspace layout (floats), with lifetime-based reuse:
static constexpr size_t MF   = 1048576;
static constexpr size_t O_XZ  = 0;               // 16M  xz1 -> xz2
static constexpr size_t O_XH  = O_XZ  + 16 * MF; //  8M  xh1 -> xh2
static constexpr size_t O_DBC = O_XH  +  8 * MF; // .5M  dbc1 -> dbc2
static constexpr size_t O_DT  = O_DBC + MF / 2;  //  8M  dt1 -> dt2
static constexpr size_t O_YMB = O_DT  +  8 * MF; //  8M  ymb1 -> ymb2
static constexpr size_t O_OUT = O_YMB +  8 * MF; //  4M  out1 -> out2
static constexpr size_t O_X2  = O_OUT +  4 * MF; //  2M  residual (persistent)
static constexpr size_t O_R7  = O_X2  +  2 * MF; //  4M  wq -> hs -> aq
static constexpr size_t O_R8  = O_R7  +  4 * MF; //  4M  xb -> xm -> g/xg
static constexpr size_t O_R9  = O_R8  +  4 * MF; //  4M  xw -> xsp -> v
static constexpr size_t O_SM  = O_R9  +  4 * MF; //  small scalars

extern "C" void kernel_launch(void* const* d_in, const int* in_sizes, int n_in,
                              void* d_out, int out_size, void* d_ws, size_t ws_size,
                              hipStream_t stream)
{
  (void)in_sizes; (void)n_in; (void)ws_size;
  const float* x = (const float*)d_in[0];
  const float* m1[9]; const float* m2[9];
  for (int i = 0; i < 9; ++i) m1[i] = (const float*)d_in[1 + i];
  for (int i = 0; i < 9; ++i) m2[i] = (const float*)d_in[10 + i];
  const float* se_w  = (const float*)d_in[19];
  const float* se_b  = (const float*)d_in[20];
  const float* bn_w  = (const float*)d_in[21];
  const float* lq_wx = (const float*)d_in[22];
  const float* lq_wh = (const float*)d_in[23];
  const float* lq_b  = (const float*)d_in[24];
  const float* lq_tau= (const float*)d_in[25];
  const float* sp_w  = (const float*)d_in[26];
  const float* sp_b  = (const float*)d_in[27];
  const float* mb_wg = (const float*)d_in[28];
  const float* mb_bg = (const float*)d_in[29];
  const float* mb_wv = (const float*)d_in[30];
  const float* mb_bv = (const float*)d_in[31];
  const float* fc_w  = (const float*)d_in[32];
  const float* fc_b  = (const float*)d_in[33];

  float* W   = (float*)d_ws;
  float* xz  = W + O_XZ;
  float* xh  = W + O_XH;
  float* dbc = W + O_DBC;
  float* dt  = W + O_DT;
  float* ymb = W + O_YMB;
  float* out = W + O_OUT;
  float* x2  = W + O_X2;   // residual
  float* R7  = W + O_R7;   // wq -> hs -> aq
  float* R8  = W + O_R8;   // xb -> xm -> g
  float* R9  = W + O_R9;   // xw -> xsp -> v
  float* part  = W + O_SM;        // K*64 partials
  float* scale = part + Kk * 64;  // K scales
  float* outp  = (float*)d_out;

  // ---- ternary quantization of bn_w (forward value) ----
  bn_partial_kernel<<<Kk * 64, 256, 0, stream>>>(bn_w, part);
  bn_scale_kernel<<<Kk, 64, 0, stream>>>(part, scale);
  bn_quant_kernel<<<(size_t)Kk * Ld * Ld / 256, 256, 0, stream>>>(bn_w, scale, R7);

  // ---- mamba 1 (b = 2) ----
  run_mamba(x, Bb, m1, xz, xh, dbc, dt, ymb, out, stream);

  // ---- squeeze/excite projection + gelu -> residual x2 ----
  gemm(out, se_w, se_b, x2, Bb * Ld, Dd, Dd, Dd, Dd, Dd, 0, A_GELU, stream);

  // ---- bn einsum: xb[k,b,m,d] = sum_l wq[k][l,m] * x2[b][l,d] ----
  for (int k = 0; k < Kk; ++k)
    for (int b = 0; b < Bb; ++b)
      gemm(R7 + (size_t)k * Ld * Ld,            // A = wq[k], transposed access
           x2 + (size_t)b * Ld * Dd,            // B = x2[b]
           nullptr,
           R8 + (size_t)(k * Bb + b) * Ld * Dd, // C = xb[k,b]
           Ld, Dd, Ld, Ld, Dd, Dd, /*transA=*/1, A_NONE, stream);

  // ---- liquid: precompute xw = xb @ wx[k] + b[k]; then sequential scan ----
  for (int k = 0; k < Kk; ++k)
    gemm(R8 + (size_t)k * Bb * Ld * Dd, lq_wx + (size_t)k * Dd * Dd,
         lq_b + (size_t)k * Dd, R9 + (size_t)k * Bb * Ld * Dd,
         Bb * Ld, Dd, Dd, Dd, Dd, Dd, 0, A_NONE, stream);
  liquid_scan_kernel<<<Kk, 1024, 0, stream>>>(R9, lq_wh, lq_tau, R7);  // hs -> R7

  // ---- top-k(256) sparsify + sp projection + silu ----
  topk_mask_kernel<<<Kk * Bb * Ld, 256, 0, stream>>>(R7, R8);          // xm -> R8
  gemm(R8, sp_w, sp_b, R9, Kk * Bb * Ld, Dd, Dd, Dd, Dd, Dd, 0, A_SILU, stream);

  // ---- mamba 2 (b = 4) ----
  run_mamba(R9, Kk * Bb, m2, xz, xh, dbc, dt, ymb, out, stream);

  // ---- act_quant ----
  actq_kernel<<<Kk * Bb * Ld, 256, 0, stream>>>(out, R7);              // aq -> R7

  // ---- gates, fc, residual ----
  gemm(R7, mb_wg, mb_bg, R8, Kk * Bb * Ld, Dd, Dd, Dd, Dd, Dd, 0, A_SIGMOID, stream);
  gemm(R7, mb_wv, mb_bv, R9, Kk * Bb * Ld, Dd, Dd, Dd, Dd, Dd, 0, A_NONE, stream);
  combine_kernel<<<(size_t)Kk * Bb * Ld * Dd / 256, 256, 0, stream>>>(R7, R8, R9);
  gemm(R8, fc_w, fc_b, outp, Kk * Bb * Ld, Dd, Dd, Dd, Dd, Dd, 0, A_NONE, stream);
  residual_add_kernel<<<(size_t)out_size / 256, 256, 0, stream>>>(outp, x2);
}
